// ClassAwareTCRLoss_12807592477410
// MI455X (gfx1250) — matrix operand order repaired and checked by reading, not compiled, not run on hardware
//
#include <hip/hip_runtime.h>
#include <hip/hip_bf16.h>
#include <math.h>

// ---- problem constants (match reference config) ----
#define DIMD 256
#define NCLS 8
#define NB   2
#define NG   16              // NB * NCLS
#define EPS2 0.04f           // 0.2^2
#define LSTR 40              // LDS k-stride (halves) per column: 80B -> bank-spread
#define AROWS 128            // output rows per workgroup (d1 block)
#define BCOLS 64             // output cols per workgroup (d2 block)

typedef __attribute__((ext_vector_type(16))) _Float16 v16h;
typedef __attribute__((ext_vector_type(8)))  float    v8f;

union V16H { v16h v; _Float16 f[16]; };
union V8F  { v8f  v; float    f[8];  };
union Pack { float4 f4; _Float16 h[8]; };

#if __has_builtin(__builtin_amdgcn_global_load_async_to_lds_b32)
#define HAVE_ASYNC 1
typedef __attribute__((address_space(1))) int* gptr_i32;
typedef __attribute__((address_space(3))) int* lptr_i32;
__device__ __forceinline__ void async_idx_b32(const int* g, int* l) {
    __builtin_amdgcn_global_load_async_to_lds_b32(
        (gptr_i32)g, (lptr_i32)l, 0, 0);
}
__device__ __forceinline__ void wait_async0() {
#if __has_builtin(__builtin_amdgcn_s_wait_asynccnt)
    __builtin_amdgcn_s_wait_asynccnt(0);
#else
    asm volatile("s_wait_asynccnt 0x0" ::: "memory");
#endif
}
#endif

// -------- kernel 0: zero meta buffers and the dummy zero row Zh[N] --------
__global__ void k_init(int* counts, int* cursors, _Float16* Zh, int N) {
    int t = threadIdx.x;
    if (t < NG) { counts[t] = 0; cursors[t] = 0; }
    Zh[(size_t)N * DIMD + t] = (_Float16)0.0f;   // row N = all zeros (pad target)
}

// -------- kernel 1: row-normalize feat -> fp16 Zh, count group sizes --------
__global__ __launch_bounds__(DIMD)
void k_normalize(const float* __restrict__ feat, const int* __restrict__ target,
                 const int* __restrict__ batch, _Float16* __restrict__ Zh,
                 int* __restrict__ counts) {
    const int row = blockIdx.x;
    const int tid = threadIdx.x;
    __shared__ float red[DIMD];
    __shared__ float sinv;

    float x = feat[row * DIMD + tid];
    red[tid] = x * x;
    __syncthreads();
    for (int s = DIMD / 2; s > 0; s >>= 1) {
        if (tid < s) red[tid] += red[tid + s];
        __syncthreads();
    }
    if (tid == 0) {
        float nrm = sqrtf(red[0]);
        sinv = 1.0f / fmaxf(nrm, 1e-12f);
        int t = target[row];
        t = t < 0 ? 0 : (t > NCLS - 1 ? NCLS - 1 : t);
        int g = batch[row] * NCLS + t;
        atomicAdd(&counts[g], 1);
    }
    __syncthreads();
    Zh[(size_t)row * DIMD + tid] = (_Float16)(x * sinv);
}

// -------- kernel 2: padded (mult-of-32) exclusive scan of group sizes --------
__global__ void k_scan(const int* __restrict__ counts, int* __restrict__ offP) {
    if (threadIdx.x == 0) {
        int acc = 0;
        for (int g = 0; g < NG; ++g) { offP[g] = acc; acc += (counts[g] + 31) & ~31; }
        offP[NG] = acc;
    }
}

// -------- kernel 3: scatter row indices into per-group padded buckets --------
__global__ void k_scatter(const int* __restrict__ target, const int* __restrict__ batch,
                          const int* __restrict__ offP, int* __restrict__ cursors,
                          int* __restrict__ idx, int N) {
    int i = blockIdx.x * blockDim.x + threadIdx.x;
    if (i >= N) return;
    int t = target[i];
    t = t < 0 ? 0 : (t > NCLS - 1 ? NCLS - 1 : t);
    int g = batch[i] * NCLS + t;
    int slot = atomicAdd(&cursors[g], 1);
    idx[offP[g] + slot] = i;
}

// -------- kernel 3b: point pad slots at the dummy zero row --------
__global__ void k_pad(const int* __restrict__ counts, const int* __restrict__ offP,
                      int* __restrict__ idx, int N) {
    int g = blockIdx.x, t = threadIdx.x;
    int n = counts[g];
    int nP = (n + 31) & ~31;
    if (n + t < nP) idx[offP[g] + n + t] = N;   // dummy zero row
}

// -------- kernel 4: grouped Gram via WMMA, LDS-staged, branch-free --------
// grid = (d2 block, d1 block, group); 256 thr = 8 waves; 128x64 output block/WG.
// Wave w owns row-strip w*16..w*16+15 and ALL four 16-col tiles (A reused 4x).
// Next chunk's 32 row indices are double-buffered into LDS via async load-to-LDS.
__global__ __launch_bounds__(256)
void k_gram(const _Float16* __restrict__ Zh, const int* __restrict__ idxP,
            const int* __restrict__ counts, const int* __restrict__ offP,
            float* __restrict__ gram) {
    const int d2b = blockIdx.x, d1b = blockIdx.y, g = blockIdx.z;
    const int tid  = threadIdx.x;
    const int w    = tid >> 5;
    const int lane = tid & 31;
    const int hi   = lane >> 4;
    const int lo   = lane & 15;

    const int n    = counts[g];
    const int nP   = (n + 31) & ~31;
    const int base = offP[g];

    // LDS tiles stored TRANSPOSED: [col][k 0..31], k-stride LSTR halves
    __shared__ _Float16 lA[AROWS * LSTR];
    __shared__ _Float16 lB[BCOLS * LSTR];
    __shared__ int ldsIdx[2][32];

    V8F cacc[4];
    #pragma unroll
    for (int j = 0; j < 4; ++j)
        #pragma unroll
        for (int r = 0; r < 8; ++r) cacc[j].f[r] = 0.0f;

    const int kr  = tid >> 3;   // staged row 0..31 within chunk
    const int seg = tid & 7;    // 8-half column segment

    // prologue: bring chunk 0's indices into LDS (wave 0)
    if (w == 0 && nP > 0) {
#ifdef HAVE_ASYNC
        async_idx_b32(idxP + base + lane, &ldsIdx[0][lane]);
        wait_async0();
#else
        ldsIdx[0][lane] = idxP[base + lane];
#endif
    }

    int cur = 0;
    for (int k0 = 0; k0 < nP; k0 += 32, cur ^= 1) {
        const int nxt = cur ^ 1;
        __syncthreads();                       // publish ldsIdx[cur]; LDS reuse fence
        // ---- kick off async prefetch of the NEXT chunk's indices
        if (w == 0 && k0 + 32 < nP) {
#ifdef HAVE_ASYNC
            async_idx_b32(idxP + base + k0 + 32 + lane, &ldsIdx[nxt][lane]);
#else
            ldsIdx[nxt][lane] = idxP[base + k0 + 32 + lane];
#endif
        }
        // ---- stage: 32 rows x (128 A-cols + 64 B-cols), b128 loads, transposed store
        const int row = ldsIdx[cur][kr];
        const _Float16* zrow = Zh + (size_t)row * DIMD;
        Pack pa0, pa1, pb;
        pa0.f4 = *(const float4*)(zrow + d1b * AROWS + seg * 8);
        pa1.f4 = *(const float4*)(zrow + d1b * AROWS + 64 + seg * 8);
        pb.f4  = *(const float4*)(zrow + d2b * BCOLS + seg * 8);
        #pragma unroll
        for (int i = 0; i < 8; ++i) {
            lA[(seg * 8 + i) * LSTR + kr]      = pa0.h[i];
            lA[(64 + seg * 8 + i) * LSTR + kr] = pa1.h[i];
            lB[(seg * 8 + i) * LSTR + kr]      = pb.h[i];
        }
#ifdef HAVE_ASYNC
        if (w == 0 && k0 + 32 < nP) wait_async0();   // before barrier that publishes nxt
#endif
        __syncthreads();

        // ---- all fragments first (10 independent ds_load_b128), then 4 WMMAs,
        //      so the scheduler can use partial DScnt waits / co-execute.
        // A 16x32 f16 layout: lanes 0-15: K 0..7 (e0-7), 16..23 (e8-15); lanes 16-31 +8
        V16H a;
        const _Float16* pA = &lA[(w * 16 + lo) * LSTR];
        *(float4*)&a.f[0] = *(const float4*)&pA[hi * 8];
        *(float4*)&a.f[8] = *(const float4*)&pA[16 + hi * 8];
        // B 32x16 f16 layout: lanes 0-15: K 0..7 (e0-7), 16..23 (e8-15); lanes 16-31 +8
        V16H b0, b1, b2, b3;
        {
            const _Float16* pB0 = &lB[(0 * 16 + lo) * LSTR + hi * 8];
            const _Float16* pB1 = &lB[(1 * 16 + lo) * LSTR + hi * 8];
            const _Float16* pB2 = &lB[(2 * 16 + lo) * LSTR + hi * 8];
            const _Float16* pB3 = &lB[(3 * 16 + lo) * LSTR + hi * 8];
            *(float4*)&b0.f[0] = *(const float4*)&pB0[0];
            *(float4*)&b0.f[8] = *(const float4*)&pB0[16];
            *(float4*)&b1.f[0] = *(const float4*)&pB1[0];
            *(float4*)&b1.f[8] = *(const float4*)&pB1[16];
            *(float4*)&b2.f[0] = *(const float4*)&pB2[0];
            *(float4*)&b2.f[8] = *(const float4*)&pB2[16];
            *(float4*)&b3.f[0] = *(const float4*)&pB3[0];
            *(float4*)&b3.f[8] = *(const float4*)&pB3[16];
        }
        cacc[0].v = __builtin_amdgcn_wmma_f32_16x16x32_f16(
            false, a.v, false, b0.v, (short)0, cacc[0].v, false, false);
        cacc[1].v = __builtin_amdgcn_wmma_f32_16x16x32_f16(
            false, a.v, false, b1.v, (short)0, cacc[1].v, false, false);
        cacc[2].v = __builtin_amdgcn_wmma_f32_16x16x32_f16(
            false, a.v, false, b2.v, (short)0, cacc[2].v, false, false);
        cacc[3].v = __builtin_amdgcn_wmma_f32_16x16x32_f16(
            false, a.v, false, b3.v, (short)0, cacc[3].v, false, false);
    }

    float* G = gram + (size_t)g * DIMD * DIMD;
    #pragma unroll
    for (int tn = 0; tn < 4; ++tn) {
        #pragma unroll
        for (int r = 0; r < 8; ++r) {
            int m = d1b * AROWS + w * 16 + r + hi * 8;  // C/D: VGPR r -> M=r (+8 hi lanes)
            G[m * DIMD + d2b * BCOLS + tn * 16 + lo] = cacc[tn].f[r];
        }
    }
}

// -------- kernel 5: M = a*gram + (1+1e-6)I, Cholesky logdet, deficit --------
__global__ __launch_bounds__(DIMD)
void k_chol(float* __restrict__ gram, const int* __restrict__ counts,
            float* __restrict__ deficits, float* __restrict__ valids) {
    const int g = blockIdx.x;
    const int tid = threadIdx.x;
    float* M = gram + (size_t)g * DIMD * DIMD;

    const float n  = (float)counts[g];
    const float nn = fmaxf(n, 1.0f);
    const float a  = (float)DIMD / (nn * EPS2);

    for (int p = tid; p < DIMD * DIMD; p += DIMD) {
        float v = a * M[p];
        if ((p >> 8) == (p & (DIMD - 1))) v += 1.0f + 1e-6f;
        M[p] = v;
    }
    __syncthreads();

    __shared__ float col[DIMD];
    __shared__ float sdiag;
    __shared__ float slog;
    if (tid == 0) slog = 0.0f;
    __syncthreads();

    for (int j = 0; j < DIMD; ++j) {
        if (tid == 0) {
            float d = M[j * DIMD + j];
            float L = sqrtf(fmaxf(d, 1e-30f));
            sdiag = L;
            slog += 2.0f * logf(L);
        }
        __syncthreads();
        col[tid] = (tid > j) ? M[tid * DIMD + j] * (1.0f / sdiag) : 0.0f;
        __syncthreads();
        if (tid > j) {
            float li = col[tid];
            for (int k = j + 1; k <= tid; ++k)
                M[tid * DIMD + k] -= li * col[k];
        }
        __syncthreads();
    }

    if (tid == 0) {
        const float log_diag = logf(1.0f + 1e-6f + 1e-12f);
        float logdet = slog + (nn - (float)DIMD) * log_diag;  // lift DxD det to NgxNg det
        float tcr = 0.5f * logdet;
        float min_tcr = 0.5f * logf((float)DIMD);
        float def = fmaxf(min_tcr - tcr, 0.0f);
        float valid = (n >= 10.0f) ? 1.0f : 0.0f;
        deficits[g] = def * valid;
        valids[g]   = valid;
    }
}

// -------- kernel 6: final per-batch average and scale --------
__global__ void k_final(const float* __restrict__ deficits,
                        const float* __restrict__ valids, float* __restrict__ out) {
    if (threadIdx.x == 0 && blockIdx.x == 0) {
        float avg = 0.0f;
        for (int b = 0; b < NB; ++b) {
            float s = 0.0f, cv = 0.0f;
            for (int c = 0; c < NCLS; ++c) {
                s  += deficits[b * NCLS + c];
                cv += valids[b * NCLS + c];
            }
            float pb = (cv > 0.0f) ? s / fmaxf(cv, 1.0f) : 0.0f;
            avg += pb;
        }
        avg *= (1.0f / (float)NB);
        out[0] = 1.0f /*loss_weight*/ * 0.05f /*lambda_tcr*/ * avg;
    }
}

extern "C" void kernel_launch(void* const* d_in, const int* in_sizes, int n_in,
                              void* d_out, int out_size, void* d_ws, size_t ws_size,
                              hipStream_t stream) {
    (void)n_in; (void)out_size; (void)ws_size;
    // inputs (setup_inputs order): pred (N*C, unused), target (N), feat (N*D), batch (N)
    const int N = in_sizes[1];
    const int*   target = (const int*)d_in[1];
    const float* feat   = (const float*)d_in[2];
    const int*   batch  = (const int*)d_in[3];
    float* out = (float*)d_out;

    // workspace layout
    char* ws = (char*)d_ws;
    _Float16* Zh   = (_Float16*)ws;                 // (N+1) rows of D halves
    size_t off = (size_t)(N + 1) * DIMD * sizeof(_Float16);
    off = (off + 255) & ~(size_t)255;
    int*   idx     = (int*)(ws + off);              off += (size_t)(N + NG * 32 + 64) * sizeof(int);
    off = (off + 255) & ~(size_t)255;
    float* gram    = (float*)(ws + off);            off += (size_t)NG * DIMD * DIMD * sizeof(float);
    int*   counts  = (int*)(ws + off);              off += NG * sizeof(int);
    int*   cursors = (int*)(ws + off);              off += NG * sizeof(int);
    int*   offP    = (int*)(ws + off);              off += (NG + 1) * sizeof(int);
    float* deficits= (float*)(ws + off);            off += NG * sizeof(float);
    float* valids  = (float*)(ws + off);

    k_init<<<1, DIMD, 0, stream>>>(counts, cursors, Zh, N);
    k_normalize<<<N, DIMD, 0, stream>>>(feat, target, batch, Zh, counts);
    k_scan<<<1, 1, 0, stream>>>(counts, offP);
    k_scatter<<<(N + 255) / 256, 256, 0, stream>>>(target, batch, offP, cursors, idx, N);
    k_pad<<<NG, 32, 0, stream>>>(counts, offP, idx, N);
    k_gram<<<dim3(DIMD / BCOLS, DIMD / AROWS, NG), 256, 0, stream>>>(Zh, idx, counts, offP, gram);
    k_chol<<<NG, DIMD, 0, stream>>>(gram, counts, deficits, valids);
    k_final<<<1, 32, 0, stream>>>(deficits, valids, out);
}
